// KGNNModel_51196010168703
// MI455X (gfx1250) — matrix operand order repaired and checked by reading, not compiled, not run on hardware
//
#include <hip/hip_runtime.h>
#include <hip/hip_bf16.h>
#include <math.h>

// ---------------------------------------------------------------------------
// Problem constants (from the reference)
// ---------------------------------------------------------------------------
#define CC   64          // channels
#define EPSB 1e-5f

typedef __attribute__((ext_vector_type(16))) __bf16 v16bf;
typedef __attribute__((ext_vector_type(8)))  float  v8f;

__device__ __forceinline__ unsigned pk2bf(float a, float b) {   // pack 2 f32 -> 2 bf16 (RNE)
  union { float f; unsigned u; } x, y; x.f = a; y.f = b;
  unsigned ra = x.u + 0x7FFFu + ((x.u >> 16) & 1u);
  unsigned rb = y.u + 0x7FFFu + ((y.u >> 16) & 1u);
  return (ra >> 16) | (rb & 0xFFFF0000u);
}
__device__ __forceinline__ float sig_f(float x) { return 1.f / (1.f + __expf(-x)); }
__device__ __forceinline__ unsigned f2ord(float f) {
  unsigned u = __float_as_uint(f);
  return (u & 0x80000000u) ? ~u : (u | 0x80000000u);
}
__device__ __forceinline__ float ord2f(unsigned u) {
  return (u & 0x80000000u) ? __uint_as_float(u & 0x7FFFFFFFu) : __uint_as_float(~u);
}

// ---------------------------------------------------------------------------
// WMMA GEMM: Y[M x Nc] = X[M x 64] @ W[64 x Nc] (+ bias), fp32 in/out,
// bf16 WMMA f32-accumulate.  REQUIRES M % 16 == 0 (true here: 60000, 12000).
// Block = 256 threads (8 waves) covering 128 rows x 32 cols; each wave owns a
// 16x32 strip (two 16x16 WMMA tiles sharing one A fragment).
// W tile (64x32) staged in LDS as packed bf16 pairs [col][kpair], stride 36
// dwords (16B-aligned rows, conflict-free column reads).
// A fragment loads: two contiguous 8-float runs per lane -> float4 loads,
// no per-lane predication (EXEC stays all-ones for WMMA).
// Launch: grid = (ceil(M/128), Nc/32), block = 256.
// ---------------------------------------------------------------------------
#define LWS 36
__global__ __launch_bounds__(256) void gemm64_wmma(
    const float* __restrict__ X, const float* __restrict__ W,
    const float* __restrict__ bias, float* __restrict__ Y, int M, int Nc) {
  __shared__ unsigned lw[32 * LWS];          // [col 0..31][kpair 0..31]
  const int n0 = blockIdx.y * 32;

  // cooperative W-tile fill: 32 kpairs x 32 cols, coalesced along col
  for (int p = threadIdx.x; p < 32 * 32; p += 256) {
    int col = p & 31, kp = p >> 5;
    float f0 = W[(size_t)(2 * kp) * Nc + n0 + col];
    float f1 = W[(size_t)(2 * kp + 1) * Nc + n0 + col];
    lw[col * LWS + kp] = pk2bf(f0, f1);
  }
  __syncthreads();

  const int wave = threadIdx.x >> 5;
  const int lane = threadIdx.x & 31;
  const int m0 = (blockIdx.x * 8 + wave) * 16;
  if (m0 >= M) return;                       // uniform per wave
  const int half = lane >> 4;                // 0: lanes 0-15, 1: lanes 16-31
  const int l16  = lane & 15;
  const int row  = m0 + l16;                 // always < M (M % 16 == 0)

  v8f acc0 = {}, acc1 = {};
#pragma unroll
  for (int k0 = 0; k0 < 64; k0 += 32) {
    // A: lane holds row `row`, K runs {k0+8h..+7} and {k0+16+8h..+7}
    const float4* xr = (const float4*)(X + (size_t)row * 64 + k0 + 8 * half);
    float4 r0 = xr[0];                       // K offsets +0..3
    float4 r1 = xr[1];                       // +4..7
    float4 r2 = xr[4];                       // +16..19
    float4 r3 = xr[5];                       // +20..23
    union { unsigned u[8]; v16bf v; } A;
    A.u[0] = pk2bf(r0.x, r0.y); A.u[1] = pk2bf(r0.z, r0.w);
    A.u[2] = pk2bf(r1.x, r1.y); A.u[3] = pk2bf(r1.z, r1.w);
    A.u[4] = pk2bf(r2.x, r2.y); A.u[5] = pk2bf(r2.z, r2.w);
    A.u[6] = pk2bf(r3.x, r3.y); A.u[7] = pk2bf(r3.z, r3.w);
    // B: lane holds col, VGPR j packs K = k0 + 8h + 2j (+8 for j>=4), +1  -> kpair base
    const int kpb = (k0 >> 1) + 4 * half;
    union { unsigned u[8]; v16bf v; } B0, B1;
#pragma unroll
    for (int j = 0; j < 8; ++j) {
      B0.u[j] = lw[l16 * LWS + kpb + j + (j & 4)];
      B1.u[j] = lw[(16 + l16) * LWS + kpb + j + (j & 4)];
    }
    acc0 = __builtin_amdgcn_wmma_f32_16x16x32_bf16(false, A.v, false, B0.v,
                                                   (short)0, acc0, false, false);
    acc1 = __builtin_amdgcn_wmma_f32_16x16x32_bf16(false, A.v, false, B1.v,
                                                   (short)0, acc1, false, false);
  }
  const float bv0 = bias ? bias[n0 + l16] : 0.f;
  const float bv1 = bias ? bias[n0 + 16 + l16] : 0.f;
#pragma unroll
  for (int r = 0; r < 8; ++r) {
    int rr = m0 + r + 8 * half;              // always < M
    Y[(size_t)rr * Nc + n0 + l16]      = acc0[r] + bv0;
    Y[(size_t)rr * Nc + n0 + 16 + l16] = acc1[r] + bv1;
  }
}

// ---------------------------------------------------------------------------
// Edge message + scatter (fine graph, with edge gate + node-type coefficient).
// One wave per edge; lane handles channels c = lane, lane+32.
// Gate recomputed in-register: sigmoid(edge_attr[e] @ We + be)  (10 x 64).
// ---------------------------------------------------------------------------
__global__ void edge_msg_fine(const float* __restrict__ hW, const float* __restrict__ ea,
                              const float* __restrict__ We, const float* __restrict__ be,
                              const float* __restrict__ ntc, const int* __restrict__ ntype,
                              const int* __restrict__ src, const int* __restrict__ dst,
                              float* __restrict__ agg, int E_) {
  int wid = blockIdx.x * (blockDim.x >> 5) + (threadIdx.x >> 5);
  if (wid >= E_) return;
  int lane = threadIdx.x & 31;
  int s = src[wid], d = dst[wid];
  int nt = ntype[s];
  const float* er = ea + (size_t)wid * 10;
  float e[10];
#pragma unroll
  for (int j = 0; j < 10; ++j) e[j] = er[j];
#pragma unroll
  for (int t = 0; t < 2; ++t) {
    int c = lane + 32 * t;
    float g = be[c];
#pragma unroll
    for (int j = 0; j < 10; ++j) g += e[j] * We[j * CC + c];
    g = sig_f(g);
    float m = hW[(size_t)s * CC + c] * g * ntc[nt * CC + c];
    atomicAdd(&agg[(size_t)d * CC + c], m);
  }
}

// Coarse graph: plain message gather + scatter-add.
__global__ void edge_msg_plain(const float* __restrict__ hW, const int* __restrict__ src,
                               const int* __restrict__ dst, float* __restrict__ agg, int E_) {
  int wid = blockIdx.x * (blockDim.x >> 5) + (threadIdx.x >> 5);
  if (wid >= E_) return;
  int lane = threadIdx.x & 31;
  int s = src[wid], d = dst[wid];
#pragma unroll
  for (int t = 0; t < 2; ++t) {
    int c = lane + 32 * t;
    atomicAdd(&agg[(size_t)d * CC + c], hW[(size_t)s * CC + c]);
  }
}

// GRU combine: h = (1-z)*tanh(inn + r*hn) + z*h
__global__ void gru_update(const float* __restrict__ gi, const float* __restrict__ gh,
                           float* __restrict__ h, int n) {
  int idx = blockIdx.x * blockDim.x + threadIdx.x;
  if (idx >= n) return;
  int i = idx >> 6, c = idx & 63;
  const float* gir = gi + (size_t)i * 192;
  const float* ghr = gh + (size_t)i * 192;
  float r  = sig_f(gir[c]       + ghr[c]);
  float z  = sig_f(gir[64 + c]  + ghr[64 + c]);
  float nn = tanhf(gir[128 + c] + r * ghr[128 + c]);
  h[idx] = (1.f - z) * nn + z * h[idx];
}

// BatchNorm stats over rows (population mean / 1/sqrt(var+eps)); block per feature
__global__ void bn_stats(const float* __restrict__ x, int M, int F,
                         float* __restrict__ mean, float* __restrict__ rstd) {
  __shared__ float sh[256], sh2[256];
  int f = blockIdx.x;
  float s = 0.f, s2 = 0.f;
  for (int i = threadIdx.x; i < M; i += blockDim.x) {
    float v = x[(size_t)i * F + f];
    s += v; s2 += v * v;
  }
  sh[threadIdx.x] = s; sh2[threadIdx.x] = s2;
  __syncthreads();
  for (int off = 128; off > 0; off >>= 1) {
    if ((int)threadIdx.x < off) { sh[threadIdx.x] += sh[threadIdx.x + off]; sh2[threadIdx.x] += sh2[threadIdx.x + off]; }
    __syncthreads();
  }
  if (threadIdx.x == 0) {
    float m = sh[0] / (float)M;
    float var = sh2[0] / (float)M - m * m;
    mean[f] = m;
    rstd[f] = rsqrtf(fmaxf(var, 0.f) + EPSB);
  }
}

__global__ void bn_apply(float* __restrict__ x, const float* __restrict__ mean,
                         const float* __restrict__ rstd, const float* __restrict__ g,
                         const float* __restrict__ b, int M, int F, int relu) {
  int idx = blockIdx.x * blockDim.x + threadIdx.x;
  if (idx >= M * F) return;
  int f = idx % F;
  float v = (x[idx] - mean[f]) * rstd[f] * g[f] + b[f];
  x[idx] = relu ? fmaxf(v, 0.f) : v;
}

__global__ void relu_k(float* __restrict__ x, int n) {
  int idx = blockIdx.x * blockDim.x + threadIdx.x;
  if (idx < n) x[idx] = fmaxf(x[idx], 0.f);
}

// avg_pool pieces
__global__ void pool_count(const int* __restrict__ adst, float* __restrict__ cnt, int n) {
  int i = blockIdx.x * blockDim.x + threadIdx.x;
  if (i < n) atomicAdd(&cnt[adst[i]], 1.f);
}
__global__ void pool_sum(const float* __restrict__ h, const int* __restrict__ adst,
                         float* __restrict__ h2, int n) {
  int idx = blockIdx.x * blockDim.x + threadIdx.x;
  if (idx >= n * CC) return;
  int i = idx >> 6, c = idx & 63;
  atomicAdd(&h2[(size_t)adst[i] * CC + c], h[(size_t)i * CC + c]);
}
__global__ void pool_div(float* __restrict__ h2, const float* __restrict__ cnt, int n2) {
  int idx = blockIdx.x * blockDim.x + threadIdx.x;
  if (idx >= n2 * CC) return;
  h2[idx] /= fmaxf(cnt[idx >> 6], 1.f);
}

// -------------------------- set2set building blocks ------------------------
__global__ void lstm_gates(const float* __restrict__ inp, int in_dim,
                           const float* __restrict__ hprev, const float* __restrict__ w_ih,
                           const float* __restrict__ w_hh, const float* __restrict__ bias,
                           float* __restrict__ g, int Bn) {
  int b = blockIdx.x, j = threadIdx.x;      // block 256 = 4*C
  if (b >= Bn) return;
  float acc = bias[j];
  for (int k = 0; k < in_dim; ++k) acc += inp[(size_t)b * in_dim + k] * w_ih[(size_t)k * 256 + j];
  for (int k = 0; k < CC; ++k)     acc += hprev[(size_t)b * CC + k]   * w_hh[(size_t)k * 256 + j];
  g[(size_t)b * 256 + j] = acc;
}
__global__ void lstm_update(const float* __restrict__ g, float* __restrict__ h,
                            float* __restrict__ c, int Bn) {
  int idx = blockIdx.x * blockDim.x + threadIdx.x;
  if (idx >= Bn * CC) return;
  int b = idx >> 6, ch = idx & 63;
  const float* gr = g + (size_t)b * 256;
  float gi = sig_f(gr[ch]);
  float gf = sig_f(gr[64 + ch]);
  float gg = tanhf(gr[128 + ch]);
  float go = sig_f(gr[192 + ch]);
  float cn = gf * c[idx] + gi * gg;
  c[idx] = cn;
  h[idx] = go * tanhf(cn);
}
__global__ void attn_init(unsigned* __restrict__ mxo, float* __restrict__ s,
                          float* __restrict__ r, int Bn) {
  int idx = blockIdx.x * blockDim.x + threadIdx.x;
  if (idx < Bn) { mxo[idx] = 0x007FFFFFu; /* f2ord(-inf) */ s[idx] = 0.f; }
  if (idx < Bn * CC) r[idx] = 0.f;
}
__global__ void attn_e_max(const float* __restrict__ x, const float* __restrict__ q,
                           const int* __restrict__ batch, float* __restrict__ e,
                           unsigned* __restrict__ mxo, int n) {
  int i = blockIdx.x * blockDim.x + threadIdx.x;
  if (i >= n) return;
  int b = batch[i];
  const float* xr = x + (size_t)i * CC;
  const float* qr = q + (size_t)b * CC;
  float dot = 0.f;
#pragma unroll
  for (int c = 0; c < CC; ++c) dot += xr[c] * qr[c];
  e[i] = dot;
  atomicMax(&mxo[b], f2ord(dot));
}
__global__ void attn_a(const float* __restrict__ e, const int* __restrict__ batch,
                       const unsigned* __restrict__ mxo, float* __restrict__ a,
                       float* __restrict__ s, int n) {
  int i = blockIdx.x * blockDim.x + threadIdx.x;
  if (i >= n) return;
  int b = batch[i];
  float mxv = ord2f(mxo[b]);
  if (!isfinite(mxv)) mxv = 0.f;
  float av = __expf(e[i] - mxv);
  a[i] = av;
  atomicAdd(&s[b], av);
}
__global__ void attn_r(const float* __restrict__ a, const float* __restrict__ s,
                       const float* __restrict__ x, const int* __restrict__ batch,
                       float* __restrict__ r, int n) {
  int idx = blockIdx.x * blockDim.x + threadIdx.x;
  if (idx >= n * CC) return;
  int i = idx >> 6, c = idx & 63;
  int b = batch[i];
  float w = a[i] / fmaxf(s[b], 1e-16f);
  atomicAdd(&r[(size_t)b * CC + c], w * x[(size_t)i * CC + c]);
}
__global__ void qstar_k(const float* __restrict__ q, const float* __restrict__ r,
                        float* __restrict__ qstar, int Bn) {
  int idx = blockIdx.x * blockDim.x + threadIdx.x;
  if (idx >= Bn * CC) return;
  int b = idx >> 6, ch = idx & 63;
  qstar[(size_t)b * 128 + ch]      = q[idx];
  qstar[(size_t)b * 128 + 64 + ch] = r[idx];
}

// -------------------------- head kernels -----------------------------------
__global__ void concat_k(const float* __restrict__ x1, const float* __restrict__ x2,
                         float* __restrict__ z, int Bn) {
  int idx = blockIdx.x * blockDim.x + threadIdx.x;
  if (idx >= Bn * 256) return;
  int b = idx >> 8, j = idx & 255;
  z[idx] = (j < 128) ? x1[(size_t)b * 128 + j] : x2[(size_t)b * 128 + j - 128];
}
__global__ void fc_k(const float* __restrict__ Zin, const float* __restrict__ Wm,
                     const float* __restrict__ bias, float* __restrict__ Y,
                     int Bn, int in, int out) {
  int idx = blockIdx.x * blockDim.x + threadIdx.x;
  if (idx >= Bn * out) return;
  int b = idx / out, j = idx % out;
  float acc = bias[j];
  for (int k = 0; k < in; ++k) acc += Zin[(size_t)b * in + k] * Wm[(size_t)k * out + j];
  Y[idx] = acc;
}

// ---------------------------------------------------------------------------
// Host side
// ---------------------------------------------------------------------------
struct ConvP { const float *W, *We, *b_hh, *b_ih, *be, *ntc, *w_hh, *w_ih; };

static inline float* wsf(char*& p, size_t nfloats) {
  float* r = (float*)p;
  p += ((nfloats * sizeof(float) + 255) / 256) * 256;
  return r;
}

static void set2set_run(const float* xfeat, const int* batch, int n, int Bn,
                        const float* wih0, const float* whh0, const float* b0,
                        const float* wih1, const float* whh1, const float* b1,
                        float* qstar, float* h0, float* c0, float* h1, float* c1,
                        float* gbuf, float* ebuf, float* abuf, unsigned* mxo,
                        float* sbuf, float* rbuf, float* out, hipStream_t stream) {
  hipMemsetAsync(qstar, 0, (size_t)Bn * 128 * 4, stream);
  hipMemsetAsync(h0, 0, (size_t)Bn * CC * 4, stream);
  hipMemsetAsync(c0, 0, (size_t)Bn * CC * 4, stream);
  hipMemsetAsync(h1, 0, (size_t)Bn * CC * 4, stream);
  hipMemsetAsync(c1, 0, (size_t)Bn * CC * 4, stream);
  for (int step = 0; step < 5; ++step) {
    lstm_gates<<<Bn, 256, 0, stream>>>(qstar, 128, h0, wih0, whh0, b0, gbuf, Bn);
    lstm_update<<<(Bn * CC + 255) / 256, 256, 0, stream>>>(gbuf, h0, c0, Bn);
    lstm_gates<<<Bn, 256, 0, stream>>>(h0, 64, h1, wih1, whh1, b1, gbuf, Bn);
    lstm_update<<<(Bn * CC + 255) / 256, 256, 0, stream>>>(gbuf, h1, c1, Bn);
    attn_init<<<(Bn * CC + 255) / 256, 256, 0, stream>>>(mxo, sbuf, rbuf, Bn);
    attn_e_max<<<(n + 255) / 256, 256, 0, stream>>>(xfeat, h1, batch, ebuf, mxo, n);
    attn_a<<<(n + 255) / 256, 256, 0, stream>>>(ebuf, batch, mxo, abuf, sbuf, n);
    attn_r<<<(n * CC + 255) / 256, 256, 0, stream>>>(abuf, sbuf, xfeat, batch, rbuf, n);
    qstar_k<<<(Bn * CC + 255) / 256, 256, 0, stream>>>(h1, rbuf, qstar, Bn);
  }
  hipMemcpyAsync(out, qstar, (size_t)Bn * 128 * 4, hipMemcpyDeviceToDevice, stream);
}

extern "C" void kernel_launch(void* const* d_in, const int* in_sizes, int n_in,
                              void* d_out, int out_size, void* d_ws, size_t ws_size,
                              hipStream_t stream) {
  (void)n_in; (void)ws_size;
  // ---- input mapping (top-level: setup_inputs insertion order; params pytree
  //      flattened jax-style: dict keys sorted, lists/tuples in order) ----
  const float* x         = (const float*)d_in[0];
  const int*   node_type = (const int*)  d_in[1];
  const int*   edge_idx  = (const int*)  d_in[2];
  const float* edge_attr = (const float*)d_in[3];
  const int*   batch     = (const int*)  d_in[4];
  const int*   assignd   = (const int*)  d_in[5];
  const int*   edge_idx2 = (const int*)  d_in[6];
  const int*   batch2    = (const int*)  d_in[7];

  const int Nn  = in_sizes[0] / CC;       // 60000  (multiple of 16)
  const int Ee  = in_sizes[3] / 10;       // 960000
  const int E2e = in_sizes[6] / 2;        // 96000
  const int N2n = in_sizes[7];            // 12000  (multiple of 16)
  const int Bn  = out_size;               // 64

  int pi = 8;
  const float* bn_g[3]; const float* bn_b[3];
  for (int k = 0; k < 3; ++k) { bn_g[k] = (const float*)d_in[pi++]; bn_b[k] = (const float*)d_in[pi++]; }
  const float* bnfc_g[2]; const float* bnfc_b[2];
  for (int k = 0; k < 2; ++k) { bnfc_g[k] = (const float*)d_in[pi++]; bnfc_b[k] = (const float*)d_in[pi++]; }
  // convs: 3 with-edge (sorted keys: W,We,b_hh,b_ih,be,ntc,w_hh,w_ih) + 2 plain
  ConvP cv[5];
  for (int k = 0; k < 3; ++k) {
    cv[k].W    = (const float*)d_in[pi++];
    cv[k].We   = (const float*)d_in[pi++];
    cv[k].b_hh = (const float*)d_in[pi++];
    cv[k].b_ih = (const float*)d_in[pi++];
    cv[k].be   = (const float*)d_in[pi++];
    cv[k].ntc  = (const float*)d_in[pi++];
    cv[k].w_hh = (const float*)d_in[pi++];
    cv[k].w_ih = (const float*)d_in[pi++];
  }
  for (int k = 3; k < 5; ++k) {
    cv[k].W    = (const float*)d_in[pi++];
    cv[k].b_hh = (const float*)d_in[pi++];
    cv[k].b_ih = (const float*)d_in[pi++];
    cv[k].w_hh = (const float*)d_in[pi++];
    cv[k].w_ih = (const float*)d_in[pi++];
    cv[k].We = cv[k].be = cv[k].ntc = nullptr;
  }
  const float* fcW[3]; const float* fcB[3];
  for (int k = 0; k < 3; ++k) { fcW[k] = (const float*)d_in[pi++]; fcB[k] = (const float*)d_in[pi++]; }
  const float* pre_g = (const float*)d_in[pi++];
  const float* pre_b = (const float*)d_in[pi++];
  const float* s2s[2][6];
  for (int s = 0; s < 2; ++s)
    for (int l = 0; l < 6; ++l) s2s[s][l] = (const float*)d_in[pi++];

  // ---- workspace carve-up ----
  char* p = (char*)d_ws;
  float*    h     = wsf(p, (size_t)Nn * CC);
  float*    hW    = wsf(p, (size_t)Nn * CC);
  float*    agg   = wsf(p, (size_t)Nn * CC);
  float*    gi    = wsf(p, (size_t)Nn * 192);
  float*    gh    = wsf(p, (size_t)Nn * 192);
  float*    h2    = wsf(p, (size_t)N2n * CC);
  float*    cnt   = wsf(p, (size_t)N2n);
  float*    ebuf  = wsf(p, (size_t)Nn);
  float*    abuf  = wsf(p, (size_t)Nn);
  unsigned* mxo   = (unsigned*)wsf(p, (size_t)Bn);
  float*    sbuf  = wsf(p, (size_t)Bn);
  float*    rbuf  = wsf(p, (size_t)Bn * CC);
  float*    hh0   = wsf(p, (size_t)Bn * CC);
  float*    cc0   = wsf(p, (size_t)Bn * CC);
  float*    hh1   = wsf(p, (size_t)Bn * CC);
  float*    cc1   = wsf(p, (size_t)Bn * CC);
  float*    gbuf  = wsf(p, (size_t)Bn * 256);
  float*    qstar = wsf(p, (size_t)Bn * 128);
  float*    x1    = wsf(p, (size_t)Bn * 128);
  float*    x2    = wsf(p, (size_t)Bn * 128);
  float*    zbuf  = wsf(p, (size_t)Bn * 256);
  float*    z1    = wsf(p, (size_t)Bn * 128);
  float*    z2    = wsf(p, (size_t)Bn * 64);
  float*    meanb = wsf(p, 256);
  float*    rstdb = wsf(p, 256);

  const int* src  = edge_idx;
  const int* dst  = edge_idx + Ee;
  const int* src2 = edge_idx2;
  const int* dst2 = edge_idx2 + E2e;

  auto gemm = [&](const float* X, const float* Wm, const float* bias, float* Y,
                  int M, int Ncol) {
    dim3 grid((unsigned)((M + 127) / 128), (unsigned)(Ncol / 32));
    gemm64_wmma<<<grid, 256, 0, stream>>>(X, Wm, bias, Y, M, Ncol);
  };

  // ---- fine graph: 3 gated convs (L=3 rounds each) + BN + ReLU ----
  hipMemcpyAsync(h, x, (size_t)Nn * CC * 4, hipMemcpyDeviceToDevice, stream);
  for (int k = 0; k < 3; ++k) {
    for (int l = 0; l < 3; ++l) {
      gemm(h, cv[k].W + (size_t)l * CC * CC, nullptr, hW, Nn, CC);
      hipMemsetAsync(agg, 0, (size_t)Nn * CC * 4, stream);
      edge_msg_fine<<<(Ee + 7) / 8, 256, 0, stream>>>(hW, edge_attr, cv[k].We, cv[k].be,
                                                      cv[k].ntc, node_type, src, dst, agg, Ee);
      gemm(agg, cv[k].w_ih, cv[k].b_ih, gi, Nn, 192);
      gemm(h,   cv[k].w_hh, cv[k].b_hh, gh, Nn, 192);
      gru_update<<<(Nn * CC + 255) / 256, 256, 0, stream>>>(gi, gh, h, Nn * CC);
    }
    bn_stats<<<CC, 256, 0, stream>>>(h, Nn, CC, meanb, rstdb);
    bn_apply<<<(Nn * CC + 255) / 256, 256, 0, stream>>>(h, meanb, rstdb, bn_g[k], bn_b[k], Nn, CC, 1);
  }

  // ---- set2set #1 over fine nodes ----
  set2set_run(h, batch, Nn, Bn, s2s[0][0], s2s[0][1], s2s[0][2],
              s2s[0][3], s2s[0][4], s2s[0][5],
              qstar, hh0, cc0, hh1, cc1, gbuf, ebuf, abuf, mxo, sbuf, rbuf, x1, stream);

  // ---- avg_pool fine -> coarse ----
  hipMemsetAsync(cnt, 0, (size_t)N2n * 4, stream);
  hipMemsetAsync(h2, 0, (size_t)N2n * CC * 4, stream);
  pool_count<<<(Nn + 255) / 256, 256, 0, stream>>>(assignd, cnt, Nn);
  pool_sum<<<(Nn * CC + 255) / 256, 256, 0, stream>>>(h, assignd, h2, Nn);
  pool_div<<<(N2n * CC + 255) / 256, 256, 0, stream>>>(h2, cnt, N2n);

  // ---- coarse graph: 2 plain gated convs + ReLU (reusing big buffers) ----
  for (int k = 3; k < 5; ++k) {
    for (int l = 0; l < 3; ++l) {
      gemm(h2, cv[k].W + (size_t)l * CC * CC, nullptr, hW, N2n, CC);
      hipMemsetAsync(agg, 0, (size_t)N2n * CC * 4, stream);
      edge_msg_plain<<<(E2e + 7) / 8, 256, 0, stream>>>(hW, src2, dst2, agg, E2e);
      gemm(agg, cv[k].w_ih, cv[k].b_ih, gi, N2n, 192);
      gemm(h2,  cv[k].w_hh, cv[k].b_hh, gh, N2n, 192);
      gru_update<<<(N2n * CC + 255) / 256, 256, 0, stream>>>(gi, gh, h2, N2n * CC);
    }
    relu_k<<<(N2n * CC + 255) / 256, 256, 0, stream>>>(h2, N2n * CC);
  }

  // ---- set2set #2 over coarse nodes ----
  set2set_run(h2, batch2, N2n, Bn, s2s[1][0], s2s[1][1], s2s[1][2],
              s2s[1][3], s2s[1][4], s2s[1][5],
              qstar, hh0, cc0, hh1, cc1, gbuf, ebuf, abuf, mxo, sbuf, rbuf, x2, stream);

  // ---- head: concat -> pre_bn -> fc/bn/relu -> fc -> bn/relu -> fc ----
  concat_k<<<(Bn * 256 + 255) / 256, 256, 0, stream>>>(x1, x2, zbuf, Bn);
  bn_stats<<<256, 256, 0, stream>>>(zbuf, Bn, 256, meanb, rstdb);
  bn_apply<<<(Bn * 256 + 255) / 256, 256, 0, stream>>>(zbuf, meanb, rstdb, pre_g, pre_b, Bn, 256, 0);

  fc_k<<<(Bn * 128 + 255) / 256, 256, 0, stream>>>(zbuf, fcW[0], fcB[0], z1, Bn, 256, 128);
  bn_stats<<<128, 256, 0, stream>>>(z1, Bn, 128, meanb, rstdb);
  bn_apply<<<(Bn * 128 + 255) / 256, 256, 0, stream>>>(z1, meanb, rstdb, bnfc_g[0], bnfc_b[0], Bn, 128, 1);

  fc_k<<<(Bn * 64 + 255) / 256, 256, 0, stream>>>(z1, fcW[1], fcB[1], z2, Bn, 128, 64);
  bn_stats<<<64, 256, 0, stream>>>(z2, Bn, 64, meanb, rstdb);
  bn_apply<<<(Bn * 64 + 255) / 256, 256, 0, stream>>>(z2, meanb, rstdb, bnfc_g[1], bnfc_b[1], Bn, 64, 1);

  fc_k<<<(Bn + 255) / 256, 256, 0, stream>>>(z2, fcW[2], fcB[2], (float*)d_out, Bn, 64, 1);
}
